// HASSBlock_60713657697115
// MI455X (gfx1250) — matrix operand order repaired and checked by reading, not compiled, not run on hardware
//
#include <hip/hip_runtime.h>
#include <hip/hip_bf16.h>

// ---------------------------------------------------------------------------
// CDNA5 WMMA (wave32): v_wmma_f32_16x16x32_f16
//   D(16x16 f32, 8 VGPR) = A(16x32 f16) x B(32x16 f16) + C
// ---------------------------------------------------------------------------
typedef _Float16 f16;
typedef __attribute__((ext_vector_type(16))) _Float16 v16h;
typedef __attribute__((ext_vector_type(8)))  float    v8f;

#define WMMA_F32_F16(a, b, c) \
    __builtin_amdgcn_wmma_f32_16x16x32_f16(false, (a), false, (b), (short)0, (c), false, false)

// ---------------------------------------------------------------------------
// gfx1250 async global->LDS copy (ASYNCcnt-tracked), with safe fallback.
// Builtin prototype (from compiler diagnostic): b128 payload is v4i
// (int vector_size(16)); global ptr in AS1, LDS ptr in AS3.
// ---------------------------------------------------------------------------
#if defined(__has_builtin)
#  if __has_builtin(__builtin_amdgcn_global_load_async_to_lds_b128)
#    define HAVE_ASYNC_LDS 1
#  endif
#endif

typedef int v4i __attribute__((vector_size(16)));
typedef __attribute__((address_space(1))) v4i* gv4i_p;
typedef __attribute__((address_space(3))) v4i* lv4i_p;

__device__ inline void cp_async16(void* lds, const void* g) {
#ifdef HAVE_ASYNC_LDS
    __builtin_amdgcn_global_load_async_to_lds_b128((gv4i_p)(g), (lv4i_p)(lds), 0, 0);
#else
    *(uint4*)lds = *(const uint4*)g;
#endif
}

__device__ inline void wait_async0() {
#if defined(__has_builtin) && __has_builtin(__builtin_amdgcn_s_wait_asynccnt)
    __builtin_amdgcn_s_wait_asynccnt(0);
#else
    asm volatile("s_wait_asynccnt 0x0" ::: "memory");
#endif
}

// A-matrix 16x32 (MxK) fragment from LDS tile [m][k], row stride ld.
// ISA 7.12.2: lanes 0-15 -> M=lane, VGPR0..3 => K=0..7, VGPR4..7 => K=16..23;
//             lanes 16-31 -> K offset +8 within each 16-block.
__device__ inline v16h frag_a16x32(const f16* t, int ld) {
    const int lane = threadIdx.x & 31;
    const int hf   = lane >> 4;
    const f16* row = t + (lane & 15) * ld;
    v16h f;
#pragma unroll
    for (int v = 0; v < 8; ++v) {
        int k = ((v < 4) ? (2 * v) : (16 + 2 * (v - 4))) + hf * 8;
        f[2 * v]     = row[k];
        f[2 * v + 1] = row[k + 1];
    }
    return f;
}

// B-matrix 32x16 (KxN) fragment from LDS tile stored [n][k], row stride ld.
// Lanes 0-15 -> N=lane, VGPR0..3 => K=0..7, VGPR4..7 => K=16..23; lanes 16-31 -> K offset +8.
__device__ inline v16h frag_b32x16(const f16* t, int ld) {
    const int lane = threadIdx.x & 31;
    const int hf   = lane >> 4;
    const f16* row = t + (lane & 15) * ld;
    v16h f;
#pragma unroll
    for (int v = 0; v < 8; ++v) {
        int k = ((v < 4) ? (2 * v) : (16 + 2 * (v - 4))) + hf * 8;
        f[2 * v]     = row[k];
        f[2 * v + 1] = row[k + 1];
    }
    return f;
}

// ---------------------------------------------------------------------------
// fp32 -> f16 conversion (one-shot staging; n % 4 == 0)
// ---------------------------------------------------------------------------
__global__ void cvt_f16_kernel(const float* __restrict__ in, f16* __restrict__ out, int n) {
    const int g = (blockIdx.x * blockDim.x + threadIdx.x) * 4;
    if (g >= n) return;
    const float4 v = *(const float4*)(in + g);
    f16 o[4] = {(f16)v.x, (f16)v.y, (f16)v.z, (f16)v.w};
    *(uint2*)(out + g) = *(const uint2*)o;
}

// ---------------------------------------------------------------------------
// GEMM: C[M,N] = epilogue(A[M,K] @ W[K,N] + bias [+ residual])
// f16 inputs, f16 WMMA, fp32 accumulation; double-buffered LDS with async
// A-tile copies. Block 256 (8 waves); tile 128x128x32; wave 64x32 (4x2 tiles).
// Requires M%128==0, N%128==0, K%32==0 (true for all shapes here).
// ---------------------------------------------------------------------------
template <int ACT, bool OUTF16, bool RES>
__global__ __launch_bounds__(256)
void gemm_wmma_kernel(const f16* __restrict__ A, const f16* __restrict__ W,
                      const float* __restrict__ bias, const float* __restrict__ res,
                      float* __restrict__ C, f16* __restrict__ C16,
                      int M, int N, int K) {
    __shared__ f16 As[2][128 * 32];   // [m][k]
    __shared__ f16 Bs[2][128 * 32];   // [n][k] (transposed store)

    const int tid = threadIdx.x;
    const int m0  = blockIdx.y * 128;
    const int n0  = blockIdx.x * 128;
    const int w   = tid >> 5;
    const int wm  = (w >> 2) * 64;
    const int wn  = (w & 3) * 32;

    const int arow = tid >> 1, acb = (tid & 1) * 16;   // A tile: 16 halves/thread
    const int bkk  = tid >> 3, bnb = (tid & 7) * 16;   // W tile: 16 halves/thread

    auto load_tiles = [&](int buf, int k0) {
        // A tile 128x32 f16: two async b128 copies per thread
        const f16* ga = A + (size_t)(m0 + arow) * K + k0 + acb;
        f16* la = &As[buf][arow * 32 + acb];
        cp_async16(la, ga);
        cp_async16(la + 8, ga + 8);
        // W tile 32x128 f16 -> transposed LDS [n][k]
        const f16* src = W + (size_t)(k0 + bkk) * N + n0 + bnb;
        f16* lb = &Bs[buf][0];
#pragma unroll
        for (int i = 0; i < 16; ++i) lb[(bnb + i) * 32 + bkk] = src[i];
        if (k0 + 32 < K)   // gfx1250 global_prefetch_b8 for the tile after next
            __builtin_prefetch(W + (size_t)(k0 + 32 + bkk) * N + n0 + bnb, 0, 0);
    };

    v8f acc[4][2];
    const v8f vzero = {};
#pragma unroll
    for (int i = 0; i < 4; ++i)
#pragma unroll
        for (int j = 0; j < 2; ++j) acc[i][j] = vzero;

    load_tiles(0, 0);
    wait_async0();
    __syncthreads();

    int cur = 0;
    for (int k0 = 0; k0 < K; k0 += 32) {
        const int nxt = cur ^ 1;
        if (k0 + 32 < K) load_tiles(nxt, k0 + 32);   // overlap with WMMA below

        v16h af[4], bf[2];
#pragma unroll
        for (int i = 0; i < 4; ++i) af[i] = frag_a16x32(&As[cur][(wm + i * 16) * 32], 32);
#pragma unroll
        for (int j = 0; j < 2; ++j) bf[j] = frag_b32x16(&Bs[cur][(wn + j * 16) * 32], 32);
#pragma unroll
        for (int i = 0; i < 4; ++i)
#pragma unroll
            for (int j = 0; j < 2; ++j)
                acc[i][j] = WMMA_F32_F16(af[i], bf[j], acc[i][j]);

        wait_async0();        // this wave's async copies into nxt are in LDS
        __syncthreads();      // everyone done reading cur / writing nxt
        cur = nxt;
    }

    // C/D layout: element (M = r + 8*(lane>=16), N = lane&15) in VGPR r.
    const int lane = tid & 31, hf = lane >> 4, nidx = lane & 15;
#pragma unroll
    for (int i = 0; i < 4; ++i)
#pragma unroll
        for (int j = 0; j < 2; ++j) {
            const int n = n0 + wn + j * 16 + nidx;
            const float bv = bias[n];
#pragma unroll
            for (int r = 0; r < 8; ++r) {
                const int m = m0 + wm + i * 16 + r + 8 * hf;
                float v = acc[i][j][r] + bv;
                if (ACT == 1) v = 0.5f * v * (1.0f + erff(v * 0.70710678118f));
                if (RES) v += res[(size_t)m * N + n];
                if (OUTF16) C16[(size_t)m * N + n] = (f16)v;
                else        C[(size_t)m * N + n] = v;
            }
        }
}

// ---------------------------------------------------------------------------
// Row LayerNorm (biased variance, eps 1e-5). One row per block.
// rowlen % blockDim.x == 0, rowlen/blockDim.x <= 16. fp32 math internally.
// ---------------------------------------------------------------------------
template <typename TI, typename TO>
__global__ void ln_kernel(const TI* __restrict__ in, TO* __restrict__ out,
                          const float* __restrict__ g, const float* __restrict__ b,
                          int rowlen) {
    __shared__ float s1[256], s2[256];
    const int row = blockIdx.x;
    const int t = threadIdx.x, bd = blockDim.x;
    const int per = rowlen / bd;
    const TI* rp = in + (size_t)row * rowlen;
    float vals[16];
    float sum = 0.f, sq = 0.f;
    for (int i = 0; i < per; ++i) {
        float v = (float)rp[i * bd + t];
        vals[i] = v; sum += v; sq += v * v;
    }
    s1[t] = sum; s2[t] = sq;
    __syncthreads();
    for (int st = bd >> 1; st > 0; st >>= 1) {
        if (t < st) { s1[t] += s1[t + st]; s2[t] += s2[t + st]; }
        __syncthreads();
    }
    const float mean = s1[0] / (float)rowlen;
    const float var  = s2[0] / (float)rowlen - mean * mean;
    const float rs   = rsqrtf(var + 1e-5f);
    TO* op = out + (size_t)row * rowlen;
    for (int i = 0; i < per; ++i) {
        const int c = i * bd + t;
        op[c] = (TO)((vals[i] - mean) * rs * g[c] + b[c]);
    }
}

// ---------------------------------------------------------------------------
// Windowed causal attention (flash-style, WMMA). One wave per (qtile, head, b).
// q/k/v layout [B,S,NH,DH] f16 (post head-LN for q,k). Window: j<=i, i-j<=256.
// Note: S%32==0 and jb<=S-32, so all chunk rows are in-bounds (no guards).
// ---------------------------------------------------------------------------
__global__ __launch_bounds__(32)
void attn_wmma_kernel(const f16* __restrict__ Q, const f16* __restrict__ Kx,
                      const f16* __restrict__ V, f16* __restrict__ O) {
    constexpr int S = 2048, NH = 16, DH = 64, WIN = 256, STR = NH * DH;
    __shared__ f16 Qs[16 * 64];   // [m][d]
    __shared__ f16 Ks[32 * 64];   // [j][d]  (B layout [n][k] for QK^T)
    __shared__ f16 Vs[64 * 32];   // [d][j]  (B layout [n][k] for PV)
    __shared__ f16 Ps[16 * 32];   // [m][j]  (A layout for PV)

    const int lane = threadIdx.x;
    const int qt = blockIdx.x, h = blockIdx.y, b = blockIdx.z;
    const int i0 = qt * 16;
    const size_t base = ((size_t)b * S) * STR + (size_t)h * DH;

    {   // Q tile: lane -> (row = lane&15, half-row seg), 4 async b128 copies
        const int r = lane & 15, seg = lane >> 4;
        const f16* qsrc = Q + base + (size_t)(i0 + r) * STR + seg * 32;
        f16* qdst = Qs + r * 64 + seg * 32;
#pragma unroll
        for (int i = 0; i < 4; ++i) cp_async16(qdst + 8 * i, qsrc + 8 * i);
    }
    wait_async0();
    __syncthreads();
    const v16h qa0 = frag_a16x32(Qs, 64);
    const v16h qa1 = frag_a16x32(Qs + 32, 64);

    const v8f vzero = {};
    v8f o[4];
#pragma unroll
    for (int t = 0; t < 4; ++t) o[t] = vzero;
    float mrow[8], lrow[8];
#pragma unroll
    for (int r = 0; r < 8; ++r) { mrow[r] = -1e30f; lrow[r] = 0.f; }

    const int hf = lane >> 4, nidx = lane & 15;
    int jstart = i0 - WIN; if (jstart < 0) jstart = 0; jstart &= ~31;

    for (int jb = jstart; jb <= i0 + 15; jb += 32) {
        __syncthreads();
        {   // K/V chunk: one key row per lane (always in-bounds, see note)
            const int jg = jb + lane;
            const f16* ksrc = Kx + base + (size_t)jg * STR;
            const f16* vsrc = V + base + (size_t)jg * STR;
            f16* kdst = Ks + lane * 64;
#pragma unroll
            for (int i = 0; i < 8; ++i) cp_async16(kdst + 8 * i, ksrc + 8 * i);
#pragma unroll
            for (int d = 0; d < 64; ++d) Vs[d * 32 + lane] = vsrc[d];   // transpose
        }
        wait_async0();
        __syncthreads();

        v8f s[2];
#pragma unroll
        for (int jt = 0; jt < 2; ++jt) {
            v8f st = vzero;
            st = WMMA_F32_F16(qa0, frag_b32x16(Ks + jt * 16 * 64, 64), st);
            st = WMMA_F32_F16(qa1, frag_b32x16(Ks + jt * 16 * 64 + 32, 64), st);
            s[jt] = st;
        }

        // mask + scale (1/sqrt(64)), online softmax
#pragma unroll
        for (int r = 0; r < 8; ++r) {
            const int i = i0 + r + 8 * hf;
            float mx = -1e30f;
#pragma unroll
            for (int jt = 0; jt < 2; ++jt) {
                const int j = jb + jt * 16 + nidx;
                float v = s[jt][r] * 0.125f;
                const bool ok = (j <= i) && (i - j <= WIN);
                v = ok ? v : -1e30f;
                s[jt][r] = v;
                mx = fmaxf(mx, v);
            }
#pragma unroll
            for (int msk = 1; msk <= 8; msk <<= 1) mx = fmaxf(mx, __shfl_xor(mx, msk, 32));
            const float nm = fmaxf(mrow[r], mx);
            float rsum = 0.f;
#pragma unroll
            for (int jt = 0; jt < 2; ++jt) {
                const float v = s[jt][r];
                const float p = (v <= -1e29f) ? 0.f : __expf(v - nm);
                rsum += p;
                Ps[(r + 8 * hf) * 32 + jt * 16 + nidx] = (f16)p;
            }
#pragma unroll
            for (int msk = 1; msk <= 8; msk <<= 1) rsum += __shfl_xor(rsum, msk, 32);
            const float alpha = (mrow[r] <= -1e29f) ? 0.f : __expf(mrow[r] - nm);
            lrow[r] = lrow[r] * alpha + rsum;
            mrow[r] = nm;
#pragma unroll
            for (int t = 0; t < 4; ++t) o[t][r] *= alpha;
        }
        __syncthreads();

        const v16h pa = frag_a16x32(Ps, 32);
#pragma unroll
        for (int t = 0; t < 4; ++t)
            o[t] = WMMA_F32_F16(pa, frag_b32x16(Vs + t * 16 * 32, 32), o[t]);
    }

#pragma unroll
    for (int t = 0; t < 4; ++t)
#pragma unroll
        for (int r = 0; r < 8; ++r) {
            const float v = o[t][r] / lrow[r];
            O[base + (size_t)(i0 + r + 8 * hf) * STR + t * 16 + nidx] = (f16)v;
        }
}

// ---------------------------------------------------------------------------
// Final routing combine: out = h + base*wm + adaptive*(1-wm)
// ---------------------------------------------------------------------------
__global__ void combine_kernel(const float* __restrict__ h, const f16* __restrict__ base,
                               const f16* __restrict__ a0, const f16* __restrict__ a1,
                               const f16* __restrict__ xn, const float* __restrict__ wm,
                               const int* __restrict__ idx, float* __restrict__ out,
                               int total, int dm) {
    const int g = blockIdx.x * blockDim.x + threadIdx.x;
    if (g >= total) return;
    const int tok = g / dm;
    const float w = wm[tok];
    const int id = idx[tok];
    const float ad = (id == 0) ? (float)a0[g] : ((id == 1) ? (float)a1[g] : (float)xn[g]);
    out[g] = h[g] + (float)base[g] * w + ad * (1.0f - w);
}

// ---------------------------------------------------------------------------
extern "C" void kernel_launch(void* const* d_in, const int* in_sizes, int n_in,
                              void* d_out, int out_size, void* d_ws, size_t ws_size,
                              hipStream_t stream) {
    (void)in_sizes; (void)n_in; (void)out_size; (void)ws_size;
    constexpr int B = 2, S = 2048, DM = 1024, NH = 16, DH = 64, DFF = 4096;
    constexpr int MT = B * S;                  // 4096 token rows

    const float* x      = (const float*)d_in[0];
    const float* wmul   = (const float*)d_in[1];
    const float* wq     = (const float*)d_in[2];
    const float* bq     = (const float*)d_in[3];
    const float* wk     = (const float*)d_in[4];
    const float* bk     = (const float*)d_in[5];
    const float* wv     = (const float*)d_in[6];
    const float* bv     = (const float*)d_in[7];
    const float* wo     = (const float*)d_in[8];
    const float* bo     = (const float*)d_in[9];
    const float* lnq_g  = (const float*)d_in[10];
    const float* lnq_b  = (const float*)d_in[11];
    const float* lnk_g  = (const float*)d_in[12];
    const float* lnk_b  = (const float*)d_in[13];
    const float* fc1_w  = (const float*)d_in[14];
    const float* fc1_b  = (const float*)d_in[15];
    const float* fc2_w  = (const float*)d_in[16];
    const float* fc2_b  = (const float*)d_in[17];
    const float* lni_g  = (const float*)d_in[18];
    const float* lni_b  = (const float*)d_in[19];
    const float* lnh_g  = (const float*)d_in[20];
    const float* lnh_b  = (const float*)d_in[21];
    const float* a256w1 = (const float*)d_in[22];
    const float* a256b1 = (const float*)d_in[23];
    const float* a256w2 = (const float*)d_in[24];
    const float* a256b2 = (const float*)d_in[25];
    const float* a512w1 = (const float*)d_in[26];
    const float* a512b1 = (const float*)d_in[27];
    const float* a512w2 = (const float*)d_in[28];
    const float* a512b2 = (const float*)d_in[29];
    const int*   widx   = (const int*)d_in[30];
    float* out = (float*)d_out;

    // --------------------------- workspace carve ---------------------------
    const size_t SEG = (size_t)MT * DM;        // 4M elements
    char* wsb = (char*)d_ws;
    size_t off = 0;
    auto carve_f32 = [&](size_t n) { float* p = (float*)(wsb + off); off += n * 4; return p; };
    auto carve_f16 = [&](size_t n) { f16*   p = (f16*)(wsb + off);   off += n * 2; return p; };

    float* hbuf  = carve_f32(SEG);             // fp32 residual trunk h
    f16* x16     = carve_f16(SEG);             // reused as a1h after qkv
    f16* q16     = carve_f16(SEG);             // reused as xn16 after attention
    f16* k16     = carve_f16(SEG);             // reused as g0h after attention
    f16* v16     = carve_f16(SEG);             // reused as g1h after attention
    f16* attn16  = carve_f16(SEG);             // reused as a0h after wo GEMM
    f16* hid16   = carve_f16(4 * SEG);         // MT x DFF
    f16* base16  = carve_f16(SEG);
    f16* wq16    = carve_f16((size_t)DM * DM);
    f16* wk16    = carve_f16((size_t)DM * DM);
    f16* wv16    = carve_f16((size_t)DM * DM);
    f16* wo16    = carve_f16((size_t)DM * DM);
    f16* fc1w16  = carve_f16((size_t)DM * DFF);
    f16* fc2w16  = carve_f16((size_t)DFF * DM);
    f16* a2w1_16 = carve_f16((size_t)DM * 256);
    f16* a2w2_16 = carve_f16((size_t)256 * DM);
    f16* a5w1_16 = carve_f16((size_t)DM * 512);
    f16* a5w2_16 = carve_f16((size_t)512 * DM);

    f16* xn16 = q16;
    f16* g0h  = k16;
    f16* g1h  = v16;
    f16* a0h  = attn16;
    f16* a1h  = x16;

    const dim3 blk(256);
    #define GRID(N_) dim3((N_) / 128, MT / 128)
    #define CVT(SRC, DST, N_) \
        cvt_f16_kernel<<<((N_) / 4 + 255) / 256, 256, 0, stream>>>((SRC), (DST), (int)(N_))

    // 0) one-shot f32 -> f16 staging (activations + all weights)
    CVT(x, x16, SEG);
    CVT(wq, wq16, (size_t)DM * DM);   CVT(wk, wk16, (size_t)DM * DM);
    CVT(wv, wv16, (size_t)DM * DM);   CVT(wo, wo16, (size_t)DM * DM);
    CVT(fc1_w, fc1w16, (size_t)DM * DFF);
    CVT(fc2_w, fc2w16, (size_t)DFF * DM);
    CVT(a256w1, a2w1_16, (size_t)DM * 256);
    CVT(a256w2, a2w2_16, (size_t)256 * DM);
    CVT(a512w1, a5w1_16, (size_t)DM * 512);
    CVT(a512w2, a5w2_16, (size_t)512 * DM);

    // 1) QKV projections (f16 out)
    gemm_wmma_kernel<0, true, false><<<GRID(DM), blk, 0, stream>>>(x16, wq16, bq, nullptr, nullptr, q16, MT, DM, DM);
    gemm_wmma_kernel<0, true, false><<<GRID(DM), blk, 0, stream>>>(x16, wk16, bk, nullptr, nullptr, k16, MT, DM, DM);
    gemm_wmma_kernel<0, true, false><<<GRID(DM), blk, 0, stream>>>(x16, wv16, bv, nullptr, nullptr, v16, MT, DM, DM);

    // 2) per-head-dim LayerNorm on q, k (rows of DH, in place, f16)
    ln_kernel<f16, f16><<<MT * NH, DH, 0, stream>>>(q16, q16, lnq_g, lnq_b, DH);
    ln_kernel<f16, f16><<<MT * NH, DH, 0, stream>>>(k16, k16, lnk_g, lnk_b, DH);

    // 3) windowed causal attention (f16 in/out, fp32 softmax + accum)
    attn_wmma_kernel<<<dim3(S / 16, NH, B), dim3(32), 0, stream>>>(q16, k16, v16, attn16);

    // 4) output projection + fp32 residual -> h (fp32 trunk)
    gemm_wmma_kernel<0, false, true><<<GRID(DM), blk, 0, stream>>>(attn16, wo16, bo, x, hbuf, nullptr, MT, DM, DM);

    // 5) xn = LN(h) -> f16
    ln_kernel<float, f16><<<MT, 256, 0, stream>>>(hbuf, xn16, lni_g, lni_b, DM);

    // 6) hid = LN(gelu(xn @ fc1 + b)) (f16 chain)
    gemm_wmma_kernel<1, true, false><<<GRID(DFF), blk, 0, stream>>>(xn16, fc1w16, fc1_b, nullptr, nullptr, hid16, MT, DFF, DM);
    ln_kernel<f16, f16><<<MT, 256, 0, stream>>>(hid16, hid16, lnh_g, lnh_b, DFF);

    // 7) base = hid @ fc2 + b
    gemm_wmma_kernel<0, true, false><<<GRID(DM), blk, 0, stream>>>(hid16, fc2w16, fc2_b, nullptr, nullptr, base16, MT, DM, DFF);

    // 8) width adapters
    gemm_wmma_kernel<1, true, false><<<GRID(256), blk, 0, stream>>>(xn16, a2w1_16, a256b1, nullptr, nullptr, g0h, MT, 256, DM);
    gemm_wmma_kernel<0, true, false><<<GRID(DM), blk, 0, stream>>>(g0h, a2w2_16, a256b2, nullptr, nullptr, a0h, MT, DM, 256);
    gemm_wmma_kernel<1, true, false><<<GRID(512), blk, 0, stream>>>(xn16, a5w1_16, a512b1, nullptr, nullptr, g1h, MT, 512, DM);
    gemm_wmma_kernel<0, true, false><<<GRID(DM), blk, 0, stream>>>(g1h, a5w2_16, a512b2, nullptr, nullptr, a1h, MT, DM, 512);

    // 9) routing combine -> fp32 output
    const int total = (int)(MT * (size_t)DM);
    combine_kernel<<<(total + 255) / 256, 256, 0, stream>>>(hbuf, base16, a0h, a1h, xn16,
                                                            wmul, widx, out, total, DM);
    #undef CVT
    #undef GRID
}
